// Window_MSA_31954556682293
// MI455X (gfx1250) — matrix operand order repaired and checked by reading, not compiled, not run on hardware
//
#include <hip/hip_runtime.h>
#include <hip/hip_bf16.h>

typedef __attribute__((ext_vector_type(16))) _Float16 v16h;
typedef __attribute__((ext_vector_type(8)))  float    v8f;

#define WMMA_F16(A,B,Cacc) \
  __builtin_amdgcn_wmma_f32_16x16x32_f16(false,(A),false,(B),(short)0,(Cacc),false,false)

// ---------------------------------------------------------------------------
// Constants for this problem instance (match reference): B=8, H=W=128, C=256,
// NH=8, WS=8, shift=4. Windowed row space M = B*nW*S = 131072.
// ---------------------------------------------------------------------------

// Kernel A: fused (cyclic-roll + window-partition + QKV projection).
// Writes q,k as (B,nW,NH,S,c) f16 and v transposed as (B,nW,NH,c,S) f16.
// grid = (2048, 8), block = 128 (4 waves). Each wave: one 16-row tile,
// two 16-col tiles, for all of Q/K/V -> 6 accumulators, 48 WMMAs.
__global__ __launch_bounds__(128)
void qkv_win_kernel(const float* __restrict__ x,
                    const float* __restrict__ Wq, const float* __restrict__ bq,
                    const float* __restrict__ Wk, const float* __restrict__ bk,
                    const float* __restrict__ Wv, const float* __restrict__ bv,
                    _Float16* __restrict__ qw, _Float16* __restrict__ kw,
                    _Float16* __restrict__ vw)
{
    // 32 weight columns x 256 K, transposed, f16, padded row stride 264.
    __shared__ _Float16 lW[3][32 * 264];

    const int n0 = blockIdx.y << 5;                 // first of 32 cols
    for (int t = threadIdx.x; t < 32 * 256; t += 128) {
        const int j = t & 31, k = t >> 5;           // coalesced over j
        lW[0][j * 264 + k] = (_Float16)Wq[k * 256 + n0 + j];
        lW[1][j * 264 + k] = (_Float16)Wk[k * 256 + n0 + j];
        lW[2][j * 264 + k] = (_Float16)Wv[k * 256 + n0 + j];
    }
    __syncthreads();

    const int wave = threadIdx.x >> 5;
    const int lane = threadIdx.x & 31;
    const int half = lane >> 4;                     // 0: lanes 0-15, 1: 16-31
    const int l16  = lane & 15;

    const long m0 = (((long)blockIdx.x << 2) + wave) << 4;   // 16-row tile base
    const long r  = m0 + l16;                       // this lane's A row
    const int b   = (int)(r >> 14);                 // / (nW*S)
    const int rem = (int)(r & 16383);
    const int wi  = rem >> 6;                       // window index
    const int s   = rem & 63;                       // index within window
    const int wh  = wi >> 4, ww = wi & 15;
    // cyclic roll by (-4,-4) fused into the source address
    const int ysrc = ((wh << 3) + (s >> 3) + 124) & 127;
    const int xsrc = ((ww << 3) + (s & 7) + 124) & 127;
    const float* xrow = x + ((((long)b << 14) + ysrc * 128 + xsrc) << 8);

    v8f aq0 = {}, ak0 = {}, av0 = {};
    v8f aq1 = {}, ak1 = {}, av1 = {};

    for (int kk = 0; kk < 256; kk += 32) {
        // A operand: 16x32 f16; contiguous 16-K run per lane
        v16h a;
        const float* pa = xrow + kk + half * 16;
#pragma unroll
        for (int i = 0; i < 8; ++i) {
            a[i]     = (_Float16)pa[i];
            a[8 + i] = (_Float16)pa[8 + i];
        }
        // B operands from LDS: contiguous 16-K run per lane
#pragma unroll
        for (int ntb = 0; ntb < 2; ++ntb) {
            const int bo = (ntb * 16 + l16) * 264 + kk + half * 16;
            v16h wqv, wkv, wvv;
#pragma unroll
            for (int i = 0; i < 16; ++i) {
                wqv[i] = lW[0][bo + i];
                wkv[i] = lW[1][bo + i];
                wvv[i] = lW[2][bo + i];
            }
            if (ntb == 0) {
                aq0 = WMMA_F16(a, wqv, aq0);
                ak0 = WMMA_F16(a, wkv, ak0);
                av0 = WMMA_F16(a, wvv, av0);
            } else {
                aq1 = WMMA_F16(a, wqv, aq1);
                ak1 = WMMA_F16(a, wkv, ak1);
                av1 = WMMA_F16(a, wvv, av1);
            }
        }
    }

    // n0 is a multiple of 32 -> one head per block column-slice
    const int head = n0 >> 5;
    const long hb  = (((((long)b << 8) + wi) << 3) + head) << 11;  // *2048
    const int s0   = (int)(m0 & 63);

#pragma unroll
    for (int ntb = 0; ntb < 2; ++ntb) {
        const float bqs = bq[n0 + ntb * 16 + l16];
        const float bks = bk[n0 + ntb * 16 + l16];
        const float bvs = bv[n0 + ntb * 16 + l16];
        const int cc = ntb * 16 + l16;
        const v8f vq = ntb ? aq1 : aq0;
        const v8f vk = ntb ? ak1 : ak0;
        const v8f vv = ntb ? av1 : av0;
#pragma unroll
        for (int i = 0; i < 8; ++i) {
            const int ss = s0 + i + half * 8;       // C layout: M = i (+8 hi half)
            qw[hb + ss * 32 + cc] = (_Float16)(vq[i] + bqs);
            kw[hb + ss * 32 + cc] = (_Float16)(vk[i] + bks);
            vw[hb + cc * 64 + ss] = (_Float16)(vv[i] + bvs);   // transposed (c x S)
        }
    }
}

// ---------------------------------------------------------------------------
// Kernel B: one wave = full 64x64 attention for one (b, window, head).
// scores = qk^T/sqrt(32) + bias, multiplicative region mask, softmax,
// out = attn @ v. grid = 16384, block = 32.
// ---------------------------------------------------------------------------
__global__ __launch_bounds__(32)
void win_attn_kernel(const _Float16* __restrict__ qw,
                     const _Float16* __restrict__ kw,
                     const _Float16* __restrict__ vw,
                     const float* __restrict__ bias_table,
                     float* __restrict__ out)
{
    __shared__ _Float16 attnL[16 * 64];             // one 16-row attn strip, f16

    const int blk = blockIdx.x;                     // ((b*nW + wi)*NH + h)
    const int h   = blk & 7;
    const int bwi = blk >> 3;
    const int wi  = bwi & 255;
    const int b   = bwi >> 8;
    const int wh  = wi >> 4, ww = wi & 15;
    const long base = (long)blk << 11;              // *2048 = S*c

    const int lane = threadIdx.x & 31;
    const int half = lane >> 4, l16 = lane & 15;

    // K as B operand (B[K=c][N=key]); k rows are contiguous 32 f16
    v16h kB[4];
#pragma unroll
    for (int nt = 0; nt < 4; ++nt) {
        const _Float16* p = kw + base + (nt * 16 + l16) * 32 + half * 16;
#pragma unroll
        for (int i = 0; i < 16; ++i) kB[nt][i] = p[i];
    }
    // V^T as B operand (B[K=key][N=cc]); vw stored (c x S) so runs are contiguous
    v16h vB[2][2];
#pragma unroll
    for (int ntc = 0; ntc < 2; ++ntc) {
        const _Float16* p0 = vw + base + (ntc * 16 + l16) * 64;
#pragma unroll
        for (int kt = 0; kt < 2; ++kt) {
            const _Float16* p = p0 + kt * 32 + half * 16;
#pragma unroll
            for (int i = 0; i < 16; ++i) vB[kt][ntc][i] = p[i];
        }
    }

    const float scale = 0.17677669529663687f;       // 1/sqrt(32)

    for (int mt = 0; mt < 4; ++mt) {
        // Q strip as A operand
        v16h qA;
        const _Float16* pq = qw + base + (mt * 16 + l16) * 32;
#pragma unroll
        for (int i = 0; i < 8; ++i) {
            qA[i]     = pq[half * 16 + i];
            qA[8 + i] = pq[half * 16 + 8 + i];
        }

        v8f sc[4];
#pragma unroll
        for (int nt = 0; nt < 4; ++nt) {
            v8f z = {};
            sc[nt] = WMMA_F16(qA, kB[nt], z);
        }

        // scale + relative-position bias + multiplicative region mask
#pragma unroll
        for (int nt = 0; nt < 4; ++nt) {
            const int key = nt * 16 + l16;
            const int ky = key >> 3, kx = key & 7;
            const int regk = (((wh == 15) && (ky >= 4)) ? 2 : 0) +
                             (((ww == 15) && (kx >= 4)) ? 1 : 0);
#pragma unroll
            for (int i = 0; i < 8; ++i) {
                const int qr = mt * 16 + half * 8 + i;
                const int qy = qr >> 3, qx = qr & 7;
                const int regq = (((wh == 15) && (qy >= 4)) ? 2 : 0) +
                                 (((ww == 15) && (qx >= 4)) ? 1 : 0);
                const int ridx = (qy - ky + 7) * 15 + (qx - kx + 7);
                const float bia = bias_table[ridx * 8 + h];
                const float v = sc[nt][i] * scale + bia;
                sc[nt][i] = (regq == regk) ? v : 0.0f;   // mult. mask, pre-softmax
            }
        }

        // softmax per row: reduce over 16 lanes (within half) x 4 tiles
#pragma unroll
        for (int i = 0; i < 8; ++i) {
            float m = fmaxf(fmaxf(sc[0][i], sc[1][i]), fmaxf(sc[2][i], sc[3][i]));
#pragma unroll
            for (int off = 1; off < 16; off <<= 1)
                m = fmaxf(m, __shfl_xor(m, off, 32));
            float ssum = 0.0f;
#pragma unroll
            for (int nt = 0; nt < 4; ++nt) {
                const float e = __expf(sc[nt][i] - m);
                sc[nt][i] = e;
                ssum += e;
            }
#pragma unroll
            for (int off = 1; off < 16; off <<= 1)
                ssum += __shfl_xor(ssum, off, 32);
            const float inv = 1.0f / ssum;
#pragma unroll
            for (int nt = 0; nt < 4; ++nt) sc[nt][i] *= inv;
        }

        // C layout -> row-major f16 strip in LDS (single wave; LDS is in-order)
#pragma unroll
        for (int nt = 0; nt < 4; ++nt)
#pragma unroll
            for (int i = 0; i < 8; ++i)
                attnL[(half * 8 + i) * 64 + nt * 16 + l16] = (_Float16)sc[nt][i];

        // out strip = attn(16x64) @ v(64x32)
        v8f o0 = {}, o1 = {};
#pragma unroll
        for (int kt = 0; kt < 2; ++kt) {
            v16h aA;
            const _Float16* pl = &attnL[l16 * 64 + kt * 32];
#pragma unroll
            for (int i = 0; i < 8; ++i) {
                aA[i]     = pl[half * 16 + i];
                aA[8 + i] = pl[half * 16 + 8 + i];
            }
            o0 = WMMA_F16(aA, vB[kt][0], o0);
            o1 = WMMA_F16(aA, vB[kt][1], o1);
        }

        // store (reference does NOT roll back)
#pragma unroll
        for (int ntc = 0; ntc < 2; ++ntc) {
            const int cc = ntc * 16 + l16;
            const v8f ov = ntc ? o1 : o0;
#pragma unroll
            for (int i = 0; i < 8; ++i) {
                const int srow = mt * 16 + half * 8 + i;
                const int y  = (wh << 3) + (srow >> 3);
                const int xg = (ww << 3) + (srow & 7);
                out[((((long)b << 14) + y * 128 + xg) << 8) + h * 32 + cc] = ov[i];
            }
        }
    }
}

extern "C" void kernel_launch(void* const* d_in, const int* in_sizes, int n_in,
                              void* d_out, int out_size, void* d_ws, size_t ws_size,
                              hipStream_t stream)
{
    const float* x  = (const float*)d_in[0];
    const float* Wq = (const float*)d_in[1];
    const float* bq = (const float*)d_in[2];
    const float* Wk = (const float*)d_in[3];
    const float* bk = (const float*)d_in[4];
    const float* Wv = (const float*)d_in[5];
    const float* bv = (const float*)d_in[6];
    const float* bias_table = (const float*)d_in[7];
    float* out = (float*)d_out;

    const long QKV = 33554432L;                      // 8*16384*256 f16 elements
    _Float16* qw = (_Float16*)d_ws;
    _Float16* kw = qw + QKV;
    _Float16* vw = kw + QKV;

    dim3 gA(2048, 8, 1);
    qkv_win_kernel<<<gA, 128, 0, stream>>>(x, Wq, bq, Wk, bk, Wv, bv, qw, kw, vw);
    win_attn_kernel<<<16384, 32, 0, stream>>>(qw, kw, vw, bias_table, out);
}
